// EPICostConstrction_56246891708869
// MI455X (gfx1250) — matrix operand order repaired and checked by reading, not compiled, not run on hardware
//
#include <hip/hip_runtime.h>

typedef __attribute__((ext_vector_type(16))) __bf16 v16bf;
typedef __attribute__((ext_vector_type(8)))  float  v8f;

// ---------------- problem constants ----------------
#define B_    2
#define C_    16
#define H_    128
#define W_    128
#define M_    9
#define ND_   9
#define MC_   144
#define KP_   160      // K padded per tap (144 -> 160)
#define HID_  18
#define OUT_  150
#define HP_   130      // H + halo
#define DP_   11       // ND + halo
#define WP_   130      // W + halo
#define CH2_  32       // hbuf channels (18 -> 32)
#define EPS_  1e-5f

// workspace element counts
#define XRP_ELEMS  ((size_t)B_*HP_*DP_*WP_*KP_)          // bf16
#define HBUF_ELEMS ((size_t)B_*HP_*DP_*WP_*CH2_)         // bf16
#define A1_ELEMS   (2*135*32*16)                          // bf16  (2 mtiles x 27*5 ksteps)
#define A2_ELEMS   (27*32*16)                             // bf16
#define A3_ELEMS   (10*5*32*16)                           // bf16
#define NTILES     (B_*H_*ND_*(W_/16))                    // 18432

__device__ __forceinline__ __bf16 f2bf(float f) {
    unsigned u = __float_as_uint(f);
    u += 0x7FFFu + ((u >> 16) & 1u);          // round-to-nearest-even
    unsigned short s = (unsigned short)(u >> 16);
    __bf16 r;
    __builtin_memcpy(&r, &s, 2);
    return r;
}
__device__ __forceinline__ unsigned pack2bf(float a, float b) {
    unsigned ua = __float_as_uint(a); ua += 0x7FFFu + ((ua >> 16) & 1u);
    unsigned ub = __float_as_uint(b); ub += 0x7FFFu + ((ub >> 16) & 1u);
    return (ua >> 16) | (ub & 0xffff0000u);
}

// ---------------- pack cost volume (bf16, halo+K padded, k = c*9+m) ----------------
__global__ void pack_xrp_k(const float* __restrict__ x, __bf16* __restrict__ xrp) {
    long long idx = (long long)blockIdx.x * blockDim.x + threadIdx.x;
    const long long total = (long long)B_ * H_ * ND_ * W_ * MC_;
    if (idx >= total) return;
    int k = (int)(idx % MC_); long long t = idx / MC_;
    int w = (int)(t % W_);  t /= W_;
    int d = (int)(t % ND_); t /= ND_;
    int h = (int)(t % H_);
    int b = (int)(t / H_);
    int m = k % 9, c = k / 9;
    int wsrc = w + (d - 4) * (m - 4);
    float v = 0.f;
    if (wsrc >= 0 && wsrc < W_)
        v = x[((((long long)b*C_ + c)*H_ + h)*W_ + wsrc)*M_ + m];
    long long o = ((((long long)b*HP_ + (h+1))*DP_ + (d+1))*WP_ + (w+1))*KP_ + k;
    xrp[o] = f2bf(v);
}

// ---------------- weight prep (BN folded), packed to per-lane A-fragment layout ----
// klocal for slot t, lane ln:  t + (ln>=16?16:0)
__global__ void prep_a1_k(const float* __restrict__ w1, const float* __restrict__ g1,
                          const float* __restrict__ v1, __bf16* __restrict__ a1) {
    int idx = blockIdx.x * blockDim.x + threadIdx.x;
    if (idx >= A1_ELEMS) return;
    int t = idx & 15, ln = (idx >> 4) & 31, rest = idx >> 9;
    int ks = rest % 135, mt = rest / 135;          // ks = tap*5 + j
    int row = mt*16 + (ln & 15);
    int klocal = t + (ln >= 16 ? 16 : 0);
    int tap = ks / 5, j = ks % 5;
    int kc = j*32 + klocal;
    float val = 0.f;
    if (row < HID_ && kc < MC_) {
        int m = kc % 9, c = kc / 9;
        int refch = m*C_ + c;
        float s = g1[row] * rsqrtf(v1[row] + EPS_);
        val = w1[(row*MC_ + refch)*27 + tap] * s;
    }
    a1[idx] = f2bf(val);
}

__global__ void prep_a2_k(const float* __restrict__ w2, const float* __restrict__ g2,
                          const float* __restrict__ v2, __bf16* __restrict__ a2) {
    int idx = blockIdx.x * blockDim.x + threadIdx.x;
    if (idx >= A2_ELEMS) return;
    int t = idx & 15, ln = (idx >> 4) & 31, tap = idx >> 9;
    int row = ln & 15;
    int klocal = t + (ln >= 16 ? 16 : 0);
    float val = 0.f;
    if (row < M_ && klocal < HID_) {
        float s = g2[row] * rsqrtf(v2[row] + EPS_);
        val = w2[(row*HID_ + klocal)*27 + tap] * s;
    }
    a2[idx] = f2bf(val);
}

__global__ void prep_a3_k(const float* __restrict__ clw, __bf16* __restrict__ a3) {
    int idx = blockIdx.x * blockDim.x + threadIdx.x;
    if (idx >= A3_ELEMS) return;
    int t = idx & 15, ln = (idx >> 4) & 31, rest = idx >> 9;
    int kb = rest % 5, mt = rest / 5;
    int row = mt*16 + (ln & 15);
    int klocal = t + (ln >= 16 ? 16 : 0);
    int kc = kb*32 + klocal;
    float val = 0.f;
    if (row < OUT_ && kc < MC_) {
        int refch = (kc % 9)*C_ + (kc / 9);
        val = clw[row*MC_ + refch];
    }
    a3[idx] = f2bf(val);
}

__global__ void prep_bias_k(const float* b1, const float* g1, const float* be1,
                            const float* m1, const float* v1,
                            const float* b2, const float* g2, const float* be2,
                            const float* m2, const float* v2,
                            const float* clb,
                            float* B1, float* B2, float* BCL) {
    int t = threadIdx.x;
    if (t < 32)  B1[t] = (t < HID_) ? ((b1[t]-m1[t])*(g1[t]*rsqrtf(v1[t]+EPS_)) + be1[t]) : 0.f;
    if (t < 16)  B2[t] = (t < M_)   ? ((b2[t]-m2[t])*(g2[t]*rsqrtf(v2[t]+EPS_)) + be2[t]) : 0.f;
    if (t < 160) BCL[t] = (t < OUT_) ? clb[t] : 0.f;
}

// ---------------- global attention branch (tiny) ----------------
__global__ void colsum_k(const float* __restrict__ x, float* __restrict__ cs) {
    int blk = blockIdx.x;               // (b,c,m)
    int w = threadIdx.x;
    int m = blk % 9, c = (blk / 9) % C_, b = blk / (9*C_);
    float s = 0.f;
    for (int h = 0; h < H_; ++h)
        s += x[((((long long)b*C_ + c)*H_ + h)*W_ + w)*M_ + m];
    cs[((b*C_ + c)*9 + m)*W_ + w] = s;
}

__global__ void gattn_k(const float* __restrict__ cs,
                        const float* gaw1, const float* gab1, const float* gag1,
                        const float* gabe1, const float* gam1, const float* gav1,
                        const float* gaw2, const float* gab2, const float* gag2,
                        const float* gabe2, const float* gam2, const float* gav2,
                        float* __restrict__ xg) {
    __shared__ float p[MC_];
    __shared__ float g[HID_];
    int b = blockIdx.x, t = threadIdx.x;
    if (t < MC_) {
        int c = t / 9, m = t % 9;
        const float* row = cs + ((b*C_ + c)*9 + m)*W_;
        float acc = 0.f;
        for (int d = 0; d < ND_; ++d) {
            int s = (d - 4) * (m - 4);
            int lo = s > 0 ? s : 0;
            int hi = s < 0 ? (W_ + s) : W_;
            float ss = 0.f;
            for (int w = lo; w < hi; ++w) ss += row[w];
            acc += ss;
        }
        p[m*C_ + c] = acc / ((float)H_ * (float)W_ * (float)ND_);
    }
    __syncthreads();
    if (t < HID_) {
        float acc = gab1[t];
        for (int ic = 0; ic < MC_; ++ic) acc += p[ic] * gaw1[t*MC_*27 + ic*27 + 13];
        acc = (acc - gam1[t]) * (gag1[t] * rsqrtf(gav1[t] + EPS_)) + gabe1[t];
        g[t] = acc > 0.f ? acc : 0.f;
    }
    __syncthreads();
    if (t < M_) {
        float acc = gab2[t];
        for (int ic = 0; ic < HID_; ++ic) acc += g[ic] * gaw2[t*HID_*27 + ic*27 + 13];
        acc = (acc - gam2[t]) * (gag2[t] * rsqrtf(gav2[t] + EPS_)) + gabe2[t];
        xg[b*16 + t] = acc;
    }
}

// ---------------- conv1 implicit GEMM: 32x16 tile, K = 27 taps x 160 ----------------
__global__ void __launch_bounds__(128) conv1_k(const __bf16* __restrict__ xrp,
                                               const __bf16* __restrict__ a1,
                                               const float* __restrict__ bias1,
                                               __bf16* __restrict__ hbuf) {
    int lane = threadIdx.x & 31, wid = threadIdx.x >> 5;
    int tile = blockIdx.x * 4 + wid;
    int wt = tile & 7; int t2 = tile >> 3;
    int d = t2 % ND_; t2 /= ND_;
    int h = t2 & 127; int b = t2 >> 7;
    int n = lane & 15;
    int koff = (lane >> 4) << 4;
    int w0 = wt * 16;

    v8f c0 = {0.f,0.f,0.f,0.f,0.f,0.f,0.f,0.f};
    v8f c1 = {0.f,0.f,0.f,0.f,0.f,0.f,0.f,0.f};
    const __bf16* bb = xrp + ((((long long)b*HP_ + (h+1))*DP_ + (d+1))*WP_ + (w0+1+n))*KP_;
    const __bf16* ap = a1 + (long long)lane * 16;

    for (int tap = 0; tap < 27; ++tap) {
        int oh = tap/9 - 1, ow = (tap/3)%3 - 1, od = tap%3 - 1;
        const __bf16* bt = bb + ((long long)oh*(DP_*WP_) + od*WP_ + ow)*KP_ + koff;
        #pragma unroll
        for (int j = 0; j < 5; ++j) {
            v16bf bf = *(const v16bf*)(bt + j*32);
            v16bf a0 = *(const v16bf*)(ap + (long long)((0*135 + tap*5 + j)*32)*16);
            v16bf a1f = *(const v16bf*)(ap + (long long)((1*135 + tap*5 + j)*32)*16);
            c0 = __builtin_amdgcn_wmma_f32_16x16x32_bf16(false, a0,  false, bf, (short)0, c0, false, false);
            c1 = __builtin_amdgcn_wmma_f32_16x16x32_bf16(false, a1f, false, bf, (short)0, c1, false, false);
        }
    }
    int rbase = (lane >> 4) * 8;
    __bf16* hb = hbuf + ((((long long)b*HP_ + (h+1))*DP_ + (d+1))*WP_ + (w0+1+n))*CH2_;
    #pragma unroll
    for (int v = 0; v < 8; ++v) {
        int r0 = rbase + v;
        float x0 = c0[v] + bias1[r0];
        x0 = x0 > 0.f ? x0 : 0.f;
        hb[r0] = f2bf(x0);                          // rows 0..15 all < 18
        int r1 = 16 + r0;
        if (r1 < HID_) {
            float x1 = c1[v] + bias1[r1];
            x1 = x1 > 0.f ? x1 : 0.f;
            hb[r1] = f2bf(x1);
        }
    }
}

// ------------- fused conv2 + sigmoid-weighting + conv_last (160x160 GEMM) ----------
__global__ void __launch_bounds__(128) fused_k(const __bf16* __restrict__ xrp,
                                               const __bf16* __restrict__ hbuf,
                                               const __bf16* __restrict__ a2,
                                               const __bf16* __restrict__ a3,
                                               const float* __restrict__ bias2,
                                               const float* __restrict__ biascl,
                                               const float* __restrict__ xg,
                                               float* __restrict__ out) {
    __shared__ float sw[4][144];                    // 9 x 16 sigmoid weights per wave
    int lane = threadIdx.x & 31, wid = threadIdx.x >> 5;
    int tile = blockIdx.x * 4 + wid;
    int wt = tile & 7; int t2 = tile >> 3;
    int d = t2 % ND_; t2 /= ND_;
    int h = t2 & 127; int b = t2 >> 7;
    int n = lane & 15;
    int koff = (lane >> 4) << 4;
    int w0 = wt * 16;

    // conv2: one 16x16 tile, K = 27 taps x 32
    v8f c2 = {0.f,0.f,0.f,0.f,0.f,0.f,0.f,0.f};
    const __bf16* hb = hbuf + ((((long long)b*HP_ + (h+1))*DP_ + (d+1))*WP_ + (w0+1+n))*CH2_ + koff;
    const __bf16* ap2 = a2 + (long long)lane * 16;
    for (int tap = 0; tap < 27; ++tap) {
        int oh = tap/9 - 1, ow = (tap/3)%3 - 1, od = tap%3 - 1;
        v16bf bf = *(const v16bf*)(hb + ((long long)oh*(DP_*WP_) + od*WP_ + ow)*CH2_);
        v16bf af = *(const v16bf*)(ap2 + (long long)(tap*32)*16);
        c2 = __builtin_amdgcn_wmma_f32_16x16x32_bf16(false, af, false, bf, (short)0, c2, false, false);
    }
    int rbase = (lane >> 4) * 8;
    #pragma unroll
    for (int v = 0; v < 8; ++v) {
        int r = rbase + v;
        if (r < M_) {
            float val = c2[v] + bias2[r] + xg[b*16 + r];
            val = 1.f / (1.f + __expf(-val));
            sw[wid][r*16 + n] = val;
        }
    }
    __syncthreads();

    // conv_last: D[160x16] = A3[160x160] x (wei .* cv)[160x16]
    const __bf16* cvb = xrp + ((((long long)b*HP_ + (h+1))*DP_ + (d+1))*WP_ + (w0+1+n))*KP_ + koff;
    const float* wrow = &sw[wid][0];
    v8f acc[10];
    #pragma unroll
    for (int i = 0; i < 10; ++i) acc[i] = (v8f){0.f,0.f,0.f,0.f,0.f,0.f,0.f,0.f};

    for (int kb = 0; kb < 5; ++kb) {
        uint4 lo = *(const uint4*)(cvb + kb*32);
        uint4 hi = *(const uint4*)(cvb + kb*32 + 8);
        unsigned u[8] = {lo.x, lo.y, lo.z, lo.w, hi.x, hi.y, hi.z, hi.w};
        union { v16bf v; unsigned d[8]; } bfr;
        int kstart = kb*32 + koff;
        #pragma unroll
        for (int i = 0; i < 8; ++i) {
            int k0 = kstart + 2*i;
            float f0 = __uint_as_float(u[i] << 16);
            float f1 = __uint_as_float(u[i] & 0xffff0000u);
            f0 *= wrow[(k0 % 9)*16 + n];
            f1 *= wrow[((k0 + 1) % 9)*16 + n];
            bfr.d[i] = pack2bf(f0, f1);
        }
        #pragma unroll
        for (int mt = 0; mt < 10; ++mt) {
            v16bf af = *(const v16bf*)(a3 + (long long)(((mt*5 + kb)*32) + lane)*16);
            acc[mt] = __builtin_amdgcn_wmma_f32_16x16x32_bf16(false, af, false, bfr.v, (short)0, acc[mt], false, false);
        }
    }

    // epilogue: out[b][oc][h][w][d] (fp32)
    #pragma unroll
    for (int mt = 0; mt < 10; ++mt) {
        #pragma unroll
        for (int v = 0; v < 8; ++v) {
            int oc = mt*16 + rbase + v;
            if (oc < OUT_) {
                float val = acc[mt][v] + biascl[oc];
                out[((((long long)b*OUT_ + oc)*H_ + h)*W_ + (w0 + n))*ND_ + d] = val;
            }
        }
    }
}

// ---------------- host launcher ----------------
extern "C" void kernel_launch(void* const* d_in, const int* in_sizes, int n_in,
                              void* d_out, int out_size, void* d_ws, size_t ws_size,
                              hipStream_t stream) {
    const float* x       = (const float*)d_in[0];
    const float* la_w1   = (const float*)d_in[1];
    const float* la_b1   = (const float*)d_in[2];
    const float* la_g1   = (const float*)d_in[3];
    const float* la_be1  = (const float*)d_in[4];
    const float* la_m1   = (const float*)d_in[5];
    const float* la_v1   = (const float*)d_in[6];
    const float* la_w2   = (const float*)d_in[7];
    const float* la_b2   = (const float*)d_in[8];
    const float* la_g2   = (const float*)d_in[9];
    const float* la_be2  = (const float*)d_in[10];
    const float* la_m2   = (const float*)d_in[11];
    const float* la_v2   = (const float*)d_in[12];
    const float* ga_w1   = (const float*)d_in[13];
    const float* ga_b1   = (const float*)d_in[14];
    const float* ga_g1   = (const float*)d_in[15];
    const float* ga_be1  = (const float*)d_in[16];
    const float* ga_m1   = (const float*)d_in[17];
    const float* ga_v1   = (const float*)d_in[18];
    const float* ga_w2   = (const float*)d_in[19];
    const float* ga_b2   = (const float*)d_in[20];
    const float* ga_g2   = (const float*)d_in[21];
    const float* ga_be2  = (const float*)d_in[22];
    const float* ga_m2   = (const float*)d_in[23];
    const float* ga_v2   = (const float*)d_in[24];
    const float* cl_w    = (const float*)d_in[25];
    const float* cl_b    = (const float*)d_in[26];
    float* out = (float*)d_out;
    (void)in_sizes; (void)n_in; (void)out_size; (void)ws_size;

    char* ws = (char*)d_ws;
    size_t off = 0;
    auto take = [&](size_t bytes) -> char* {
        char* p = ws + off;
        off = (off + bytes + 255) & ~(size_t)255;
        return p;
    };
    __bf16* xrp   = (__bf16*)take(XRP_ELEMS * 2);
    __bf16* hbuf  = (__bf16*)take(HBUF_ELEMS * 2);
    __bf16* a1    = (__bf16*)take(A1_ELEMS * 2);
    __bf16* a2    = (__bf16*)take(A2_ELEMS * 2);
    __bf16* a3    = (__bf16*)take(A3_ELEMS * 2);
    float*  B1    = (float*)take(32 * 4);
    float*  B2    = (float*)take(16 * 4);
    float*  BCL   = (float*)take(160 * 4);
    float*  CS    = (float*)take((size_t)B_*C_*9*W_ * 4);
    float*  XG    = (float*)take(32 * 4);

    // zero halos / padding channels (interiors are fully overwritten below)
    hipMemsetAsync(xrp,  0, XRP_ELEMS * 2,  stream);
    hipMemsetAsync(hbuf, 0, HBUF_ELEMS * 2, stream);

    // weight / bias prep
    prep_a1_k<<<A1_ELEMS/256, 256, 0, stream>>>(la_w1, la_g1, la_v1, a1);
    prep_a2_k<<<A2_ELEMS/256, 256, 0, stream>>>(la_w2, la_g2, la_v2, a2);
    prep_a3_k<<<A3_ELEMS/256, 256, 0, stream>>>(cl_w, a3);
    prep_bias_k<<<1, 256, 0, stream>>>(la_b1, la_g1, la_be1, la_m1, la_v1,
                                       la_b2, la_g2, la_be2, la_m2, la_v2,
                                       cl_b, B1, B2, BCL);

    // global attention branch
    colsum_k<<<B_*C_*9, W_, 0, stream>>>(x, CS);
    gattn_k<<<B_, 192, 0, stream>>>(CS, ga_w1, ga_b1, ga_g1, ga_be1, ga_m1, ga_v1,
                                        ga_w2, ga_b2, ga_g2, ga_be2, ga_m2, ga_v2, XG);

    // pack cost volume (bf16, halo padded)
    {
        long long total = (long long)B_ * H_ * ND_ * W_ * MC_;
        int blocks = (int)((total + 255) / 256);
        pack_xrp_k<<<blocks, 256, 0, stream>>>(x, xrp);
    }

    // conv1 (WMMA implicit GEMM) -> hbuf
    conv1_k<<<NTILES/4, 128, 0, stream>>>(xrp, a1, B1, hbuf);

    // conv2 + sigmoid + weighted conv_last (WMMA) -> out
    fused_k<<<NTILES/4, 128, 0, stream>>>(xrp, hbuf, a2, a3, B2, BCL, XG, out);
}
